// CascadedGroupAttention_chunk_31516470018779
// MI455X (gfx1250) — matrix operand order repaired and checked by reading, not compiled, not run on hardware
//
#include <hip/hip_runtime.h>

// ---------------------------------------------------------------------------
// CDNA5 (gfx1250) wave32 WMMA implementation of CascadedGroupAttention.
// All matrix work uses v_wmma_f32_16x16x32_bf16 (bf16 A/B, f32 accumulate).
// LDS tiles are stored in fragment-packed order so every WMMA operand load is
// 2x ds_load_b128; global staging uses float4 (global_load_b128); bf16
// attention tiles are staged with GLOBAL_LOAD_ASYNC_TO_LDS_B128.
// ---------------------------------------------------------------------------

typedef __attribute__((ext_vector_type(16))) __bf16 v16bf;
typedef __attribute__((ext_vector_type(8)))  float  v8f;
typedef __attribute__((ext_vector_type(4)))  int    v4i;

#if __has_builtin(__builtin_amdgcn_global_load_async_to_lds_b128) && \
    __has_builtin(__builtin_amdgcn_s_wait_asynccnt)
#define HAVE_ASYNC_LDS 1
typedef __attribute__((address_space(1))) v4i* gv4i_p;
typedef __attribute__((address_space(3))) v4i* lv4i_p;
__device__ inline void async_copy16(const void* g, void* l) {
  __builtin_amdgcn_global_load_async_to_lds_b128((gv4i_p)(g), (lv4i_p)(l), 0, 0);
}
#endif

__device__ inline unsigned short f2bf(float f) {
  union { float f; unsigned u; } v; v.f = f;
  unsigned r = v.u + 0x7FFFu + ((v.u >> 16) & 1u);   // round-to-nearest-even
  return (unsigned short)(r >> 16);
}

__device__ inline unsigned long long pack4bf(float a, float b, float c, float d) {
  return (unsigned long long)f2bf(a) | ((unsigned long long)f2bf(b) << 16) |
         ((unsigned long long)f2bf(c) << 32) | ((unsigned long long)f2bf(d) << 48);
}

union FragU { v16bf v; unsigned short s[16]; };

// A-operand, 16x32 bf16, LDS row-major [row][k], ld = 32 ushorts.
// Contiguous K order (matches packed B): lanes 0-15 hold K=0..15, lanes 16-31 hold K=16..31.
// Each half is 8 contiguous ushorts (16B, aligned) -> 2x ds_load_b128.
__device__ inline v16bf load_frag_a(const unsigned short* p, int ld) {
  int lane = threadIdx.x & 31, half = lane >> 4, r = lane & 15;
  FragU u;
#pragma unroll
  for (int i = 0; i < 8; ++i) {
    u.s[i]     = p[r * ld + half * 16 + i];
    u.s[i + 8] = p[r * ld + half * 16 + 8 + i];
  }
  return u.v;
}

// B-operand 32x16 bf16 in fragment-packed LDS: lane l's 16 elements are the
// 16 contiguous ushorts at p[l*16].  (Packed index: element(k,n) lives at
// ((n>>4)*32 + (k>>4)*16 + (n&15))*16 + (k&15).)
__device__ inline int bpack_idx(int k, int n) {
  return (((n >> 4) * 32) + ((k >> 4) * 16) + (n & 15)) * 16 + (k & 15);
}
__device__ inline v16bf load_frag_b_packed(const unsigned short* p) {
  int lane = threadIdx.x & 31;
  FragU u;
#pragma unroll
  for (int i = 0; i < 16; ++i) u.s[i] = p[lane * 16 + i];
  return u.v;
}

__device__ inline v8f zero8() {
  v8f z;
#pragma unroll
  for (int i = 0; i < 8; ++i) z[i] = 0.f;
  return z;
}

#define WMMA_BF16(a, b, c) \
  __builtin_amdgcn_wmma_f32_16x16x32_bf16(false, (a), false, (b), (short)0, (c), false, false)

// ---------------------------------------------------------------------------
// Generic tiled GEMM: OUT[b] (MxN) = A[b] (MxK f32 row-major) * X[b] (KxN f32)
// + bias[M].  8 waves arranged WM x WN; each wave owns a 64x64 tile.
// ---------------------------------------------------------------------------
template <int WM, int WN, bool RELU_IN>
__global__ __launch_bounds__(256) void gemm_bf16(
    float* __restrict__ out, const float* __restrict__ A,
    const float* __restrict__ X, const float* __restrict__ bias,
    int M, int K, int N, long strideA, long strideX, long strideO) {
  constexpr int BM = WM * 64, BN = WN * 64, BN4 = BN / 4;
  __shared__ unsigned short ldsA[BM * 32];
  __shared__ unsigned short ldsB[32 * BN];

  const int b = blockIdx.z;
  const float* Ab = A + (size_t)b * strideA;
  const float* Xb = X + (size_t)b * strideX;
  float*       Ob = out + (size_t)b * strideO;

  const int n0 = blockIdx.x * BN, m0 = blockIdx.y * BM;
  const int tid = threadIdx.x, lane = tid & 31, wv = tid >> 5;
  const int wm = (wv / WN) * 64, wn = (wv % WN) * 64;

  v8f acc[4][4];
#pragma unroll
  for (int i = 0; i < 4; ++i)
#pragma unroll
    for (int j = 0; j < 4; ++j) acc[i][j] = zero8();

  for (int kk = 0; kk < K; kk += 32) {
    // A tile: float4 global load -> 4 packed bf16 -> single 8B LDS store
    for (int t4 = tid; t4 < BM * 8; t4 += 256) {
      int row = t4 >> 3, c4 = (t4 & 7) * 4;
      const float4 av =
          *reinterpret_cast<const float4*>(Ab + (unsigned)(m0 + row) * K + kk + c4);
      *reinterpret_cast<unsigned long long*>(&ldsA[row * 32 + c4]) =
          pack4bf(av.x, av.y, av.z, av.w);
    }
    // X tile: float4 global load -> fragment-packed bf16 stores
    for (int t4 = tid; t4 < 32 * BN4; t4 += 256) {
      int row = t4 / BN4, c4 = (t4 % BN4) * 4;
      float4 xv =
          *reinterpret_cast<const float4*>(Xb + (unsigned)(kk + row) * N + n0 + c4);
      if (RELU_IN) {
        xv.x = fmaxf(xv.x, 0.f); xv.y = fmaxf(xv.y, 0.f);
        xv.z = fmaxf(xv.z, 0.f); xv.w = fmaxf(xv.w, 0.f);
      }
      ldsB[bpack_idx(row, c4 + 0)] = f2bf(xv.x);
      ldsB[bpack_idx(row, c4 + 1)] = f2bf(xv.y);
      ldsB[bpack_idx(row, c4 + 2)] = f2bf(xv.z);
      ldsB[bpack_idx(row, c4 + 3)] = f2bf(xv.w);
    }
    if (kk + 32 < K)  // speculative prefetch of next K-slab -> global_prefetch
      __builtin_prefetch(Xb + (unsigned)(kk + 32) * N + n0 + (tid % BN), 0, 1);
    __syncthreads();

    v16bf bfr[4];
#pragma unroll
    for (int ni = 0; ni < 4; ++ni)
      bfr[ni] = load_frag_b_packed(ldsB + ((wn >> 4) + ni) * 512);
#pragma unroll
    for (int mi = 0; mi < 4; ++mi) {
      v16bf a = load_frag_a(ldsA + (wm + mi * 16) * 32, 32);
#pragma unroll
      for (int ni = 0; ni < 4; ++ni) acc[mi][ni] = WMMA_BF16(a, bfr[ni], acc[mi][ni]);
    }
    __syncthreads();
  }

  const int half = lane >> 4, r = lane & 15;
#pragma unroll
  for (int mi = 0; mi < 4; ++mi)
#pragma unroll
    for (int ni = 0; ni < 4; ++ni) {
      int n = n0 + wn + ni * 16 + r;
#pragma unroll
      for (int v = 0; v < 8; ++v) {
        int m = m0 + wm + mi * 16 + v + 8 * half;
        Ob[(unsigned)m * N + n] = acc[mi][ni][v] + bias[m];
      }
    }
}

// ---------------------------------------------------------------------------
// Depthwise 5x5 conv, SAME padding, on the 128 q-channels of feat.
// ---------------------------------------------------------------------------
__global__ __launch_bounds__(256) void dwconv5(
    float* __restrict__ q, const float* __restrict__ feat,
    const float* __restrict__ w, const float* __restrict__ bias) {
  unsigned idx = blockIdx.x * 256u + threadIdx.x;    // over 32*128*4096
  int wp = idx & 63, hp = (idx >> 6) & 63, c = (idx >> 12) & 127, b = idx >> 19;
  const float* src = feat + (((unsigned)(b * 256 + c)) << 12);
  const float* kw = w + c * 25;
  float acc = bias[c];
#pragma unroll
  for (int dy = -2; dy <= 2; ++dy) {
    int y = hp + dy;
    if ((unsigned)y >= 64u) continue;
#pragma unroll
    for (int dx = -2; dx <= 2; ++dx) {
      int xw = wp + dx;
      if ((unsigned)xw >= 64u) continue;
      acc += kw[(dy + 2) * 5 + dx + 2] * src[y * 64 + xw];
    }
  }
  q[(((unsigned)(b * 128 + c)) << 12) + hp * 64 + wp] = acc;
}

// ---------------------------------------------------------------------------
// Attention scores + softmax, shared by all heads.  AXIS=0: per (b,h) over w
// (uses bias_w); AXIS=1: per (b,w) over h (uses bias_h).  2 waves/block,
// one (b,line) per wave.  K-dim = 128 channels, 4 chunks of 32.
// Output: bf16 attention matrices [b][line][q][k].
// ---------------------------------------------------------------------------
template <int AXIS>
__global__ __launch_bounds__(64) void attn_scores(
    unsigned short* __restrict__ attn, const float* __restrict__ qbuf,
    const float* __restrict__ feat, const float* __restrict__ bias_vec) {
  __shared__ unsigned short stQ[2][64 * 32];   // row-major [p][c]
  __shared__ unsigned short stK[2][32 * 64];   // fragment-packed
  __shared__ float sc[2][64 * 64];

  const int wave = threadIdx.x >> 5, lane = threadIdx.x & 31;
  const int pair = blockIdx.x * 2 + wave;
  const int b = pair >> 6, line = pair & 63;

  v8f acc[4][4];
#pragma unroll
  for (int i = 0; i < 4; ++i)
#pragma unroll
    for (int j = 0; j < 4; ++j) acc[i][j] = zero8();

  for (int c0 = 0; c0 < 128; c0 += 32) {
    if (AXIS == 0) {
      const float* qb = qbuf + (((unsigned)(b * 128 + c0)) << 12) + (unsigned)line * 64;
      const float* kb = feat + (((unsigned)(b * 256 + 128 + c0)) << 12) + (unsigned)line * 64;
      for (int t4 = lane; t4 < 512; t4 += 32) {   // 32*64/4 float4 chunks
        int c = t4 >> 4, p = (t4 & 15) * 4;
        const float4 qv = *reinterpret_cast<const float4*>(qb + ((unsigned)c << 12) + p);
        const float4 kv = *reinterpret_cast<const float4*>(kb + ((unsigned)c << 12) + p);
        stQ[wave][(p + 0) * 32 + c] = f2bf(qv.x);
        stQ[wave][(p + 1) * 32 + c] = f2bf(qv.y);
        stQ[wave][(p + 2) * 32 + c] = f2bf(qv.z);
        stQ[wave][(p + 3) * 32 + c] = f2bf(qv.w);
        stK[wave][bpack_idx(c, p + 0)] = f2bf(kv.x);
        stK[wave][bpack_idx(c, p + 1)] = f2bf(kv.y);
        stK[wave][bpack_idx(c, p + 2)] = f2bf(kv.z);
        stK[wave][bpack_idx(c, p + 3)] = f2bf(kv.w);
      }
    } else {
      for (int t = lane; t < 32 * 64; t += 32) {
        int c = t >> 6, p = t & 63;
        float qv = qbuf[(((unsigned)(b * 128 + c0 + c)) << 12) + (unsigned)p * 64 + line];
        float kv = feat[(((unsigned)(b * 256 + 128 + c0 + c)) << 12) + (unsigned)p * 64 + line];
        stQ[wave][p * 32 + c] = f2bf(qv);
        stK[wave][bpack_idx(c, p)] = f2bf(kv);
      }
    }
    __syncthreads();
    v16bf bfr[4];
#pragma unroll
    for (int ni = 0; ni < 4; ++ni) bfr[ni] = load_frag_b_packed(&stK[wave][ni * 512]);
#pragma unroll
    for (int mi = 0; mi < 4; ++mi) {
      v16bf a = load_frag_a(&stQ[wave][mi * 16 * 32], 32);
#pragma unroll
      for (int ni = 0; ni < 4; ++ni) acc[mi][ni] = WMMA_BF16(a, bfr[ni], acc[mi][ni]);
    }
    __syncthreads();
  }

  const int half = lane >> 4, r = lane & 15;
#pragma unroll
  for (int mi = 0; mi < 4; ++mi)
#pragma unroll
    for (int ni = 0; ni < 4; ++ni) {
      int k = ni * 16 + r;
#pragma unroll
      for (int v = 0; v < 8; ++v) {
        int qrow = mi * 16 + v + 8 * half;
        int d = qrow > k ? qrow - k : k - qrow;
        sc[wave][qrow * 64 + k] = acc[mi][ni][v] * 0.25f + bias_vec[d];
      }
    }
  __syncthreads();

  for (int qrow = lane; qrow < 64; qrow += 32) {
    float m = -3.4e38f;
    for (int k = 0; k < 64; ++k) m = fmaxf(m, sc[wave][qrow * 64 + k]);
    float s = 0.f;
    for (int k = 0; k < 64; ++k) {
      float e = __expf(sc[wave][qrow * 64 + k] - m);
      sc[wave][qrow * 64 + k] = e;
      s += e;
    }
    float inv = 1.f / s;
    unsigned base = (((unsigned)(b * 64 + line)) * 64 + qrow) * 64;
    for (int k = 0; k < 64; ++k) attn[base + k] = f2bf(sc[wave][qrow * 64 + k] * inv);
  }
}

// ---------------------------------------------------------------------------
// Cascade chunk accumulate: facc = x_chunk(head) [+ facc if head>0].
// ---------------------------------------------------------------------------
__global__ __launch_bounds__(256) void chunk_acc(
    float* __restrict__ facc, const float* __restrict__ x, int head) {
  unsigned i = blockIdx.x * 256u + threadIdx.x;      // over 32*64*4096
  unsigned p = i & 4095, c = (i >> 12) & 63, b = i >> 18;
  float v = x[((b * 512u + head * 64u + c) << 12) + p];
  if (head) v += facc[i];
  facc[i] = v;
}

// ---------------------------------------------------------------------------
// attn_w apply: v2[b][h][q][d] = sum_k attnW[b,h,q,k] * vbuf[b,d,h,k].
// 4 waves/block, one (b,h) each.  K-dim = 64 (2 chunks of 32).
// ---------------------------------------------------------------------------
__global__ __launch_bounds__(128) void apply_attn_w(
    float* __restrict__ v2, const unsigned short* __restrict__ attnW,
    const float* __restrict__ vbuf) {
  __shared__ unsigned short sA[4][64 * 32];
  __shared__ unsigned short sB[4][32 * 64];   // fragment-packed
  const int wave = threadIdx.x >> 5, lane = threadIdx.x & 31;
  const int pair = blockIdx.x * 4 + wave;
  const int b = pair >> 6, h = pair & 63;

  v8f acc[4][4];
#pragma unroll
  for (int i = 0; i < 4; ++i)
#pragma unroll
    for (int j = 0; j < 4; ++j) acc[i][j] = zero8();

  for (int c0 = 0; c0 < 64; c0 += 32) {
    // A: pure bf16 copy of the attention tile -> async DMA to LDS when available
    const unsigned short* ga = attnW + ((unsigned)(b * 64 + h) * 64) * 64 + c0;
#ifdef HAVE_ASYNC_LDS
    for (int cidx = lane; cidx < 256; cidx += 32) {   // 256 x 16B chunks
      int qrow = cidx >> 2, part = (cidx & 3) * 8;
      async_copy16(ga + (unsigned)qrow * 64 + part, &sA[wave][qrow * 32 + part]);
    }
    __builtin_amdgcn_s_wait_asynccnt(0);
#else
    for (int t = lane; t < 64 * 32; t += 32) {
      int qrow = t >> 5, k = t & 31;
      sA[wave][t] = ga[(unsigned)qrow * 64 + k];
    }
#endif
    // B: gather V rows (contiguous in k), store fragment-packed transposed
    for (int t4 = lane; t4 < 512; t4 += 32) {
      int d = t4 >> 3, k4 = (t4 & 7) * 4;
      const float4 vv = *reinterpret_cast<const float4*>(
          vbuf + (((unsigned)(b * 64 + d)) << 12) + (unsigned)h * 64 + c0 + k4);
      sB[wave][bpack_idx(k4 + 0, d)] = f2bf(vv.x);
      sB[wave][bpack_idx(k4 + 1, d)] = f2bf(vv.y);
      sB[wave][bpack_idx(k4 + 2, d)] = f2bf(vv.z);
      sB[wave][bpack_idx(k4 + 3, d)] = f2bf(vv.w);
    }
    __syncthreads();
    v16bf bfr[4];
#pragma unroll
    for (int ni = 0; ni < 4; ++ni) bfr[ni] = load_frag_b_packed(&sB[wave][ni * 512]);
#pragma unroll
    for (int mi = 0; mi < 4; ++mi) {
      v16bf a = load_frag_a(&sA[wave][mi * 16 * 32], 32);
#pragma unroll
      for (int ni = 0; ni < 4; ++ni) acc[mi][ni] = WMMA_BF16(a, bfr[ni], acc[mi][ni]);
    }
    __syncthreads();
  }

  const int half = lane >> 4, r = lane & 15;
#pragma unroll
  for (int mi = 0; mi < 4; ++mi)
#pragma unroll
    for (int ni = 0; ni < 4; ++ni) {
      int d = ni * 16 + r;
#pragma unroll
      for (int v = 0; v < 8; ++v) {
        int qrow = mi * 16 + v + 8 * half;
        v2[(((unsigned)(b * 64 + h)) * 64 + qrow) * 64 + d] = acc[mi][ni][v];
      }
    }
}

// ---------------------------------------------------------------------------
// attn_h apply + scatter to concat layout:
// concat[b, head*64+d, qh, w] = sum_k attnH[b,w,qh,k] * v2[b,k,w,d].
// ---------------------------------------------------------------------------
__global__ __launch_bounds__(128) void apply_attn_h(
    float* __restrict__ concat, const unsigned short* __restrict__ attnH,
    const float* __restrict__ v2, int head) {
  __shared__ unsigned short sA[4][64 * 32];
  __shared__ unsigned short sB[4][32 * 64];   // fragment-packed
  const int wave = threadIdx.x >> 5, lane = threadIdx.x & 31;
  const int pair = blockIdx.x * 4 + wave;
  const int b = pair >> 6, w = pair & 63;

  v8f acc[4][4];
#pragma unroll
  for (int i = 0; i < 4; ++i)
#pragma unroll
    for (int j = 0; j < 4; ++j) acc[i][j] = zero8();

  for (int c0 = 0; c0 < 64; c0 += 32) {
    const unsigned short* ga = attnH + ((unsigned)(b * 64 + w) * 64) * 64 + c0;
#ifdef HAVE_ASYNC_LDS
    for (int cidx = lane; cidx < 256; cidx += 32) {
      int qrow = cidx >> 2, part = (cidx & 3) * 8;
      async_copy16(ga + (unsigned)qrow * 64 + part, &sA[wave][qrow * 32 + part]);
    }
    __builtin_amdgcn_s_wait_asynccnt(0);
#else
    for (int t = lane; t < 64 * 32; t += 32) {
      int qrow = t >> 5, k = t & 31;
      sA[wave][t] = ga[(unsigned)qrow * 64 + k];
    }
#endif
    for (int t4 = lane; t4 < 512; t4 += 32) {
      int kk = t4 >> 4, d = (t4 & 15) * 4;
      const float4 vv = *reinterpret_cast<const float4*>(
          v2 + ((unsigned)((b * 64 + c0 + kk) * 64 + w)) * 64 + d);
      sB[wave][bpack_idx(kk, d + 0)] = f2bf(vv.x);
      sB[wave][bpack_idx(kk, d + 1)] = f2bf(vv.y);
      sB[wave][bpack_idx(kk, d + 2)] = f2bf(vv.z);
      sB[wave][bpack_idx(kk, d + 3)] = f2bf(vv.w);
    }
    __syncthreads();
    v16bf bfr[4];
#pragma unroll
    for (int ni = 0; ni < 4; ++ni) bfr[ni] = load_frag_b_packed(&sB[wave][ni * 512]);
#pragma unroll
    for (int mi = 0; mi < 4; ++mi) {
      v16bf a = load_frag_a(&sA[wave][mi * 16 * 32], 32);
#pragma unroll
      for (int ni = 0; ni < 4; ++ni) acc[mi][ni] = WMMA_BF16(a, bfr[ni], acc[mi][ni]);
    }
    __syncthreads();
  }

  const int half = lane >> 4, r = lane & 15;
#pragma unroll
  for (int mi = 0; mi < 4; ++mi)
#pragma unroll
    for (int ni = 0; ni < 4; ++ni) {
      int d = ni * 16 + r;
#pragma unroll
      for (int v = 0; v < 8; ++v) {
        int qh = mi * 16 + v + 8 * half;
        concat[(((unsigned)(b * 512 + head * 64 + d)) << 12) + qh * 64 + w] =
            acc[mi][ni][v];
      }
    }
}

// ---------------------------------------------------------------------------
// Host-side launch sequence.
// ---------------------------------------------------------------------------
extern "C" void kernel_launch(void* const* d_in, const int* in_sizes, int n_in,
                              void* d_out, int out_size, void* d_ws, size_t ws_size,
                              hipStream_t stream) {
  (void)in_sizes; (void)n_in; (void)out_size; (void)ws_size;
  const float* x      = (const float*)d_in[0];
  const float* qk_w   = (const float*)d_in[1];
  const float* qk_b   = (const float*)d_in[2];
  const float* dws_w  = (const float*)d_in[3];
  const float* dws_b  = (const float*)d_in[4];
  const float* vs_w   = (const float*)d_in[5];
  const float* vs_b   = (const float*)d_in[6];
  const float* proj_w = (const float*)d_in[7];
  const float* proj_b = (const float*)d_in[8];
  const float* bias_h = (const float*)d_in[9];
  const float* bias_w = (const float*)d_in[10];

  char* ws = (char*)d_ws;
  float*          feat   = (float*)(ws);                          // 32*256*4096 f32
  float*          qbuf   = (float*)(ws + 134217728);              // 32*128*4096 f32
  unsigned short* attnW  = (unsigned short*)(ws + 201326592);     // 32*64*64*64 bf16
  unsigned short* attnH  = (unsigned short*)(ws + 218103808);     // 32*64*64*64 bf16
  float*          facc   = (float*)(ws + 234881024);              // 32*64*4096 f32
  float*          vbuf   = (float*)(ws + 268435456);              // 32*64*4096 f32
  float*          v2     = (float*)(ws + 301989888);              // 32*64*64*64 f32
  float*          concat = (float*)(ws + 335544320);              // 32*512*4096 f32

  // 1) QK projection: per batch (256x512)*(512x4096) + bias
  dim3 g1(16, 2, 32);
  gemm_bf16<2, 4, false><<<g1, 256, 0, stream>>>(
      feat, qk_w, x, qk_b, 256, 512, 4096, 0L, 512L * 4096, 256L * 4096);

  // 2) depthwise 5x5 on q-channels
  dwconv5<<<(32 * 128 * 4096) / 256, 256, 0, stream>>>(qbuf, feat, dws_w, dws_b);

  // 3) attention maps (shared across heads)
  attn_scores<0><<<1024, 64, 0, stream>>>(attnW, qbuf, feat, bias_w);
  attn_scores<1><<<1024, 64, 0, stream>>>(attnH, qbuf, feat, bias_h);

  // 4) cascade over heads
  for (int i = 0; i < 8; ++i) {
    chunk_acc<<<32768, 256, 0, stream>>>(facc, x, i);
    dim3 gv(8, 1, 32);
    gemm_bf16<1, 8, false><<<gv, 256, 0, stream>>>(
        vbuf, vs_w + (size_t)i * 64 * 64, facc, vs_b + (size_t)i * 64,
        64, 64, 4096, 0L, 64L * 4096, 64L * 4096);
    apply_attn_w<<<512, 128, 0, stream>>>(v2, attnW, vbuf);
    apply_attn_h<<<512, 128, 0, stream>>>(concat, attnH, v2, i);
  }

  // 5) ReLU (fused into load) + final projection (512x512)*(512x4096) + bias
  dim3 gp(16, 4, 32);
  gemm_bf16<2, 4, true><<<gp, 256, 0, stream>>>(
      (float*)d_out, proj_w, concat, proj_b, 512, 512, 4096,
      0L, 512L * 4096, 512L * 4096);
}
